// CustomMambaModel_85220741088113
// MI455X (gfx1250) — matrix operand order repaired and checked by reading, not compiled, not run on hardware
//
#include <hip/hip_runtime.h>
#include <hip/hip_bf16.h>

// ---------------------------------------------------------------------------
// 2-layer linear RNN (scan) + final FC for MI455X (gfx1250, wave32, WMMA).
//   Prepass : x -> bf16 (xb), 5 weight matrices -> bf16, fused biases.
//   Phase 1 : A[s,b,:] = x[b,s,:] @ Wic0^T  -- WMMA GEMM, double-buffered LDS
//             tiles filled with global_load_async_to_lds_b128 (ASYNCcnt).
//   Phase 2 : 2048-step recurrence, 8 co-resident WGs, weight slices held in
//             LDS (192KB/WGP, async-preloaded), bf16 hidden state double-
//             buffered in L2, f32 WMMA accumulation + f32 A_t carry,
//             device-wide sense barrier between layer phases.
//   Phase 3 : out = h1_final @ fcW^T + fcb.
// ---------------------------------------------------------------------------

#define BATCH 64
#define SEQ   2048
#define HID   512

typedef unsigned short u16;
typedef __attribute__((ext_vector_type(16))) __bf16 v16bf;
typedef __attribute__((ext_vector_type(8)))  float  v8f;

struct __align__(16) U128 { unsigned x, y, z, w; };

union Frag {
    U128  q[2];
    v16bf v;
};

__device__ inline v8f vzero() {
    v8f z;
#pragma unroll
    for (int i = 0; i < 8; ++i) z[i] = 0.0f;
    return z;
}

// f32 -> bf16, round-to-nearest-even
__device__ inline u16 f2bf(float f) {
    unsigned u = __builtin_bit_cast(unsigned, f);
    u += 0x7FFFu + ((u >> 16) & 1u);
    return (u16)(u >> 16);
}
__device__ inline unsigned pack2(float a, float b) {
    return (unsigned)f2bf(a) | ((unsigned)f2bf(b) << 16);
}

__device__ inline v8f bfwmma(const Frag& a, const Frag& b, v8f c) {
    return __builtin_amdgcn_wmma_f32_16x16x32_bf16(false, a.v, false, b.v,
                                                   (short)0, c, false, false);
}

// --- CDNA5 async global->LDS copy (tracked by ASYNCcnt) --------------------
// lds_off: wave-relative LDS byte offset (low 32 bits of shared aperture addr)
__device__ inline void async_ld_b128(unsigned lds_off, const void* gaddr) {
    asm volatile("global_load_async_to_lds_b128 %0, %1, off"
                 :: "v"(lds_off), "v"((unsigned long long)(uintptr_t)gaddr)
                 : "memory");
}
__device__ inline void wait_async0() {
    asm volatile("s_wait_asynccnt 0" ::: "memory");
}
__device__ inline unsigned lds_off(const void* p) {
    return (unsigned)(uintptr_t)p;   // shared aperture: low 32 bits = LDS addr
}

// Device-wide sense barrier (8 co-resident workgroups).
__device__ inline void gridSync(unsigned* cnt, unsigned* gen, unsigned nwg) {
    __syncthreads();
    if (threadIdx.x == 0) {
        __threadfence();
        unsigned g = __hip_atomic_load(gen, __ATOMIC_RELAXED, __HIP_MEMORY_SCOPE_AGENT);
        unsigned a = __hip_atomic_fetch_add(cnt, 1u, __ATOMIC_ACQ_REL, __HIP_MEMORY_SCOPE_AGENT);
        if (a == nwg - 1u) {
            __hip_atomic_store(cnt, 0u, __ATOMIC_RELAXED, __HIP_MEMORY_SCOPE_AGENT);
            __hip_atomic_fetch_add(gen, 1u, __ATOMIC_ACQ_REL, __HIP_MEMORY_SCOPE_AGENT);
        } else {
            while (__hip_atomic_load(gen, __ATOMIC_ACQUIRE, __HIP_MEMORY_SCOPE_AGENT) == g) {
                __builtin_amdgcn_s_sleep(2);
            }
        }
    }
    __syncthreads();
}

// K-loop over a 512-deep GEMM slab: wave computes 16 rows x 32 cols
// (two 16x16 tiles); one A-fragment feeds both WMMAs.
// hsrc: bf16 activations, rows of 512. Wb: bf16 weights, rows of 512
// (may be an LDS pointer -> ds_load_b128 after address-space inference).
__device__ inline void gemm_slab(const u16* __restrict__ hsrc,
                                 const u16* __restrict__ Wb,
                                 unsigned m0, unsigned n0,
                                 unsigned lr, unsigned kg,
                                 v8f& acc0, v8f& acc1) {
    for (unsigned kb = 0; kb < 512; kb += 32) {
        Frag a, b0, b1;
        const u16* hr = hsrc + (size_t)(m0 + lr) * 512 + kb;
        a.q[0] = *(const U128*)(hr + kg * 16);       // K = kg*16 .. +7
        a.q[1] = *(const U128*)(hr + kg * 16 + 8);   // K = kg*16+8 .. +7
        const u16* w0 = Wb + (size_t)(n0 + lr) * 512 + kb + kg * 16;
        b0.q[0] = *(const U128*)(w0);
        b0.q[1] = *(const U128*)(w0 + 8);
        const u16* w1 = Wb + (size_t)(n0 + 16 + lr) * 512 + kb + kg * 16;
        b1.q[0] = *(const U128*)(w1);
        b1.q[1] = *(const U128*)(w1 + 8);
        acc0 = bfwmma(a, b0, acc0);
        acc1 = bfwmma(a, b1, acc1);
    }
}

// ---------------------------------------------------------------------------
// Prepass: x (f32) -> bf16, 8 elements per thread.
__global__ __launch_bounds__(256) void k_xb(const float4* __restrict__ x,
                                            U128* __restrict__ xb) {
    size_t i = (size_t)blockIdx.x * 256u + threadIdx.x;
    float4 f0 = x[2 * i], f1 = x[2 * i + 1];
    U128 o;
    o.x = pack2(f0.x, f0.y);
    o.y = pack2(f0.z, f0.w);
    o.z = pack2(f1.x, f1.y);
    o.w = pack2(f1.z, f1.w);
    xb[i] = o;
}

// Prepass: convert 5 weight matrices (512x512) to bf16.
__global__ __launch_bounds__(256) void k_prep(const float* __restrict__ Wic0,
                                              const float* __restrict__ Whc0,
                                              const float* __restrict__ Wic1,
                                              const float* __restrict__ Whc1,
                                              const float* __restrict__ fcW,
                                              u16* Wi0b, u16* Wh0b, u16* Wi1b,
                                              u16* Wh1b, u16* fcWb) {
    unsigned idx = blockIdx.x * 256u + threadIdx.x;
    unsigned mat = idx >> 18;
    unsigned off = idx & 262143u;
    const float* s;
    u16* d;
    switch (mat) {
        case 0:  s = Wic0; d = Wi0b; break;
        case 1:  s = Whc0; d = Wh0b; break;
        case 2:  s = Wic1; d = Wi1b; break;
        case 3:  s = Whc1; d = Wh1b; break;
        default: s = fcW;  d = fcWb; break;
    }
    d[off] = f2bf(s[off]);
}

__global__ __launch_bounds__(256) void k_bias(const float* bic0, const float* bhc0, const float* bc0,
                                              const float* bic1, const float* bhc1, const float* bc1,
                                              float* bias0, float* bias1) {
    unsigned i = blockIdx.x * 256u + threadIdx.x;   // grid=2 -> 512
    bias0[i] = bic0[i] + bhc0[i] + bc0[i];
    bias1[i] = bic1[i] + bhc1[i] + bc1[i];
}

__global__ __launch_bounds__(256) void k_init(U128* h0buf, U128* h1buf, unsigned* syncp) {
    unsigned i = blockIdx.x * 256u + threadIdx.x;   // grid=32 -> 8192 (x16B = 128KB each)
    U128 z = {0u, 0u, 0u, 0u};
    h0buf[i] = z;
    h1buf[i] = z;
    if (i == 0) { syncp[0] = 0u; syncp[1] = 0u; }
}

// ---------------------------------------------------------------------------
// Phase 1: A = x @ Wic0^T (bf16 operands). M=131072 N=512 K=512.
// Double-buffered LDS tiles filled with async global->LDS B128 copies.
__global__ __launch_bounds__(256) void k_xw_gemm(const u16* __restrict__ xb,
                                                 const u16* __restrict__ Wb,
                                                 float* __restrict__ Aout) {
    __shared__ u16 Xs[2][128][40];   // stride 40 keeps every 8-elem chunk 16B aligned
    __shared__ u16 Ws_[2][128][40];

    const unsigned tid  = threadIdx.x;
    const unsigned lane = tid & 31u;
    const unsigned w    = tid >> 5;
    const unsigned wm   = w & 3u;    // 4 m-groups of 32 rows
    const unsigned wn   = w >> 2;    // 2 n-groups of 64 cols
    const unsigned bm   = blockIdx.x * 128u;
    const unsigned bn   = blockIdx.y * 128u;
    const unsigned lr   = lane & 15u;
    const unsigned kg   = lane >> 4;

    v8f acc[2][4];
#pragma unroll
    for (int i = 0; i < 2; ++i)
#pragma unroll
        for (int j = 0; j < 4; ++j) acc[i][j] = vzero();

    const unsigned r  = tid >> 1;          // 0..127 tile row
    const unsigned c0 = (tid & 1u) * 2u;   // first of two 8-elem chunks

    // stage(kb, buf): async-copy one 128x32 bf16 tile of X and of W into LDS
    auto stage = [&](unsigned kb, unsigned buf) {
        const u16* gx = xb + (size_t)(bm + r) * 512 + kb;
        async_ld_b128(lds_off(&Xs[buf][r][c0 * 8]),       gx + c0 * 8);
        async_ld_b128(lds_off(&Xs[buf][r][(c0 + 1) * 8]), gx + (c0 + 1) * 8);
        const u16* gw = Wb + (size_t)(bn + r) * 512 + kb;
        async_ld_b128(lds_off(&Ws_[buf][r][c0 * 8]),       gw + c0 * 8);
        async_ld_b128(lds_off(&Ws_[buf][r][(c0 + 1) * 8]), gw + (c0 + 1) * 8);
    };

    stage(0, 0);
    for (unsigned i = 0; i < 16; ++i) {
        const unsigned buf = i & 1u;
        wait_async0();          // this wave's async copies (current+prev) done
        __syncthreads();        // => every wave's copies into buf are done
        if (i + 1 < 16) stage((i + 1) * 32, buf ^ 1u);   // overlap next copy

        Frag a[2];
#pragma unroll
        for (int mt = 0; mt < 2; ++mt) {
            const unsigned row = wm * 32 + mt * 16 + lr;
            a[mt].q[0] = *(const U128*)&Xs[buf][row][kg * 16];
            a[mt].q[1] = *(const U128*)&Xs[buf][row][kg * 16 + 8];
        }
#pragma unroll
        for (int nt = 0; nt < 4; ++nt) {
            const unsigned nr = wn * 64 + nt * 16 + lr;
            Frag b;
            b.q[0] = *(const U128*)&Ws_[buf][nr][kg * 16];
            b.q[1] = *(const U128*)&Ws_[buf][nr][kg * 16 + 8];
#pragma unroll
            for (int mt = 0; mt < 2; ++mt) acc[mt][nt] = bfwmma(a[mt], b, acc[mt][nt]);
        }
        __syncthreads();        // buf fully consumed before it is re-staged
    }

    // store: row m = b*SEQ + s  ->  Aout[(s*64 + b)*512 + n]
#pragma unroll
    for (int mt = 0; mt < 2; ++mt)
#pragma unroll
        for (int nt = 0; nt < 4; ++nt) {
            const unsigned gn = bn + wn * 64 + nt * 16 + lr;
#pragma unroll
            for (int rr = 0; rr < 8; ++rr) {
                const unsigned gm = bm + wm * 32 + mt * 16 + kg * 8 + rr;
                const unsigned bb = gm >> 11;
                const unsigned ss = gm & 2047u;
                Aout[((size_t)ss * 64 + bb) * 512 + gn] = acc[mt][nt][rr];
            }
        }
}

// ---------------------------------------------------------------------------
// Phase 2: sequential recurrence. 8 WGs x 256 threads, co-resident.
// Per-WG weight slices (3 x 64x512 bf16 = 192KB) async-preloaded into LDS.
__global__ __launch_bounds__(256) void k_recur(const float* __restrict__ A,
                                               const u16* __restrict__ Wh0,
                                               const u16* __restrict__ Wi1,
                                               const u16* __restrict__ Wh1,
                                               const float* __restrict__ bias0,
                                               const float* __restrict__ bias1,
                                               u16* h0buf, u16* h1buf,
                                               unsigned* syncp) {
    __shared__ u16 Wlds[3][64][512];   // 192KB / WGP (one WG per WGP)

    const unsigned tid  = threadIdx.x;
    const unsigned lane = tid & 31u;
    const unsigned w    = tid >> 5;
    const unsigned lr   = lane & 15u;
    const unsigned kg   = lane >> 4;
    const unsigned m0   = (w >> 1) * 16u;                 // 0..48
    const unsigned wgN  = blockIdx.x * 64u;
    const unsigned n0g  = wgN + (w & 1u) * 32u;           // global cols n0g, n0g+16
    const unsigned n0l  = (w & 1u) * 32u;                 // LDS-local cols
    const unsigned nwg  = gridDim.x;

    {   // async-preload the three 64x512 weight slices for this WG's columns
        const u16* srcs[3] = { Wh0 + (size_t)wgN * 512,
                               Wi1 + (size_t)wgN * 512,
                               Wh1 + (size_t)wgN * 512 };
        for (unsigned c = tid; c < 3u * 8192u; c += 256u) {   // 16B chunks
            const unsigned m    = c >> 13;                    // 8192 chunks/matrix
            const unsigned rest = c & 8191u;
            async_ld_b128(lds_off(((const U128*)&Wlds[m][0][0]) + rest),
                          ((const U128*)srcs[m]) + rest);
        }
        wait_async0();
        __syncthreads();
    }

    const u16* Wl0 = &Wlds[0][0][0];
    const u16* Wl1 = &Wlds[1][0][0];
    const u16* Wl2 = &Wlds[2][0][0];

    for (unsigned t = 0; t < SEQ; ++t) {
        const u16* h0p = h0buf + (size_t)(t & 1u) * (64 * 512);
        u16*       h0n = h0buf + (size_t)((t + 1u) & 1u) * (64 * 512);
        const u16* h1p = h1buf + (size_t)(t & 1u) * (64 * 512);
        u16*       h1n = h1buf + (size_t)((t + 1u) & 1u) * (64 * 512);

        // ---- layer 0: h0_new = A_t + h0_prev @ Whc0^T + bias0 ----
        v8f acc0 = vzero(), acc1 = vzero();
        gemm_slab(h0p, Wl0, m0, n0l, lr, kg, acc0, acc1);

        // prefetch next timestep's A slab (global_prefetch)
        if (t + 1u < SEQ) {
            const float* pf = A + ((size_t)(t + 1u) * 64 + (tid >> 2)) * 512 + wgN + (tid & 3u) * 16u;
            __builtin_prefetch(pf, 0, 1);
        }

        const float* At = A + (size_t)t * (64 * 512);
#pragma unroll
        for (int rr = 0; rr < 8; ++rr) {
            const unsigned m = m0 + kg * 8 + rr;
            {
                const unsigned n = n0g + lr;
                h0n[(size_t)m * 512 + n] = f2bf(acc0[rr] + At[(size_t)m * 512 + n] + bias0[n]);
            }
            {
                const unsigned n = n0g + 16 + lr;
                h0n[(size_t)m * 512 + n] = f2bf(acc1[rr] + At[(size_t)m * 512 + n] + bias0[n]);
            }
        }
        gridSync(syncp, syncp + 1, nwg);

        // ---- layer 1: h1_new = h0_new @ Wic1^T + h1_prev @ Whc1^T + bias1 ----
        acc0 = vzero(); acc1 = vzero();
        gemm_slab(h0n, Wl1, m0, n0l, lr, kg, acc0, acc1);
        gemm_slab(h1p, Wl2, m0, n0l, lr, kg, acc0, acc1);
#pragma unroll
        for (int rr = 0; rr < 8; ++rr) {
            const unsigned m = m0 + kg * 8 + rr;
            {
                const unsigned n = n0g + lr;
                h1n[(size_t)m * 512 + n] = f2bf(acc0[rr] + bias1[n]);
            }
            {
                const unsigned n = n0g + 16 + lr;
                h1n[(size_t)m * 512 + n] = f2bf(acc1[rr] + bias1[n]);
            }
        }
        gridSync(syncp, syncp + 1, nwg);
    }
}

// ---------------------------------------------------------------------------
// Phase 3: out = h1_final @ fcW^T + fcb (h1_final is parity-0 buffer).
__global__ __launch_bounds__(256) void k_fc(const u16* __restrict__ h1,
                                            const u16* __restrict__ Wfc,
                                            const float* __restrict__ fcb,
                                            float* __restrict__ out) {
    const unsigned tid  = threadIdx.x;
    const unsigned lane = tid & 31u;
    const unsigned w    = tid >> 5;
    const unsigned lr   = lane & 15u;
    const unsigned kg   = lane >> 4;
    const unsigned m0   = (w >> 1) * 16u;
    const unsigned n0   = blockIdx.x * 64u + (w & 1u) * 32u;

    v8f acc0 = vzero(), acc1 = vzero();
    gemm_slab(h1, Wfc, m0, n0, lr, kg, acc0, acc1);

#pragma unroll
    for (int rr = 0; rr < 8; ++rr) {
        const unsigned m = m0 + kg * 8 + rr;
        out[(size_t)m * 512 + (n0 + lr)]      = acc0[rr] + fcb[n0 + lr];
        out[(size_t)m * 512 + (n0 + 16 + lr)] = acc1[rr] + fcb[n0 + 16 + lr];
    }
}

// ---------------------------------------------------------------------------
extern "C" void kernel_launch(void* const* d_in, const int* in_sizes, int n_in,
                              void* d_out, int out_size, void* d_ws, size_t ws_size,
                              hipStream_t stream) {
    (void)in_sizes; (void)n_in; (void)out_size; (void)ws_size;

    const float* x    = (const float*)d_in[0];
    const float* Wic0 = (const float*)d_in[1];
    const float* bic0 = (const float*)d_in[2];
    const float* Whc0 = (const float*)d_in[3];
    const float* bhc0 = (const float*)d_in[4];
    const float* bc0  = (const float*)d_in[5];
    const float* Wic1 = (const float*)d_in[6];
    const float* bic1 = (const float*)d_in[7];
    const float* Whc1 = (const float*)d_in[8];
    const float* bhc1 = (const float*)d_in[9];
    const float* bc1  = (const float*)d_in[10];
    const float* fcW  = (const float*)d_in[11];
    const float* fcb  = (const float*)d_in[12];
    float* out = (float*)d_out;

    // workspace layout
    char* ws = (char*)d_ws;
    size_t off = 0;
    float* A = (float*)(ws + off);      off += (size_t)SEQ * BATCH * HID * 4;   // 256 MB
    u16* xb = (u16*)(ws + off);         off += (size_t)SEQ * BATCH * HID * 2;   // 128 MB
    u16* Wi0b = (u16*)(ws + off);       off += 512 * 512 * 2;
    u16* Wh0b = (u16*)(ws + off);       off += 512 * 512 * 2;
    u16* Wi1b = (u16*)(ws + off);       off += 512 * 512 * 2;
    u16* Wh1b = (u16*)(ws + off);       off += 512 * 512 * 2;
    u16* fcWb = (u16*)(ws + off);       off += 512 * 512 * 2;
    float* bias0 = (float*)(ws + off);  off += 512 * 4;
    float* bias1 = (float*)(ws + off);  off += 512 * 4;
    u16* h0buf = (u16*)(ws + off);      off += 2 * 64 * 512 * 2;
    u16* h1buf = (u16*)(ws + off);      off += 2 * 64 * 512 * 2;
    unsigned* syncp = (unsigned*)(ws + off); off += 256;

    k_xb<<<32768, 256, 0, stream>>>((const float4*)x, (U128*)xb);
    k_prep<<<5120, 256, 0, stream>>>(Wic0, Whc0, Wic1, Whc1, fcW,
                                     Wi0b, Wh0b, Wi1b, Wh1b, fcWb);
    k_bias<<<2, 256, 0, stream>>>(bic0, bhc0, bc0, bic1, bhc1, bc1, bias0, bias1);
    k_init<<<32, 256, 0, stream>>>((U128*)h0buf, (U128*)h1buf, syncp);
    k_xw_gemm<<<dim3(1024, 4, 1), 256, 0, stream>>>(xb, Wi0b, A);
    k_recur<<<8, 256, 0, stream>>>(A, Wh0b, Wi1b, Wh1b, bias0, bias1, h0buf, h1buf, syncp);
    k_fc<<<8, 256, 0, stream>>>(h1buf /* parity 0 = final */, fcWb, fcb, out);
}
